// Evaluator_80668075754095
// MI455X (gfx1250) — matrix operand order repaired and checked by reading, not compiled, not run on hardware
//
#include <hip/hip_runtime.h>
#include <hip/hip_bf16.h>
#include <math.h>

// ---------------------------------------------------------------------------
// Types / WMMA helpers (CDNA5 gfx1250, wave32, V_WMMA_F32_16X16X32_F16)
// ---------------------------------------------------------------------------
typedef _Float16 h16;
typedef __attribute__((ext_vector_type(16))) _Float16 v16h;
typedef __attribute__((ext_vector_type(8)))  _Float16 v8h;
typedef __attribute__((ext_vector_type(8)))  float    v8f;
typedef unsigned int u32x4 __attribute__((ext_vector_type(4)));
typedef int          i32x8 __attribute__((ext_vector_type(8)));
typedef int          i32x4 __attribute__((ext_vector_type(4)));

// A fragment (16x32 f16). ISA 7.12.2: lanes 0-15 / 16-31 both hold rows 0-15,
// lane half selects K sub-range. Two contiguous 16B vector loads per lane.
__device__ __forceinline__ v16h frag_a(const h16* p, int ld) {
  int lane = threadIdx.x & 31;
  const h16* r = p + (size_t)(lane & 15) * ld + ((lane >> 4) << 3);
  v8h lo = *(const v8h*)r;          // K sub-block 0
  v8h hi = *(const v8h*)(r + 16);   // K sub-block 1
  v16h a;
#pragma unroll
  for (int e = 0; e < 8; ++e) { a[e] = lo[e]; a[e + 8] = hi[e]; }
  return a;
}

// B fragment from swizzled weight block (512 halves per 32x16 K-N tile,
// layout [col][k slot], k order as in frag_a): one contiguous 32B vector load per lane.
__device__ __forceinline__ v16h frag_bs(const h16* blk) {
  int lane = threadIdx.x & 31;
  return *(const v16h*)(blk + (lane & 15) * 32 + ((lane >> 4) << 4));
}

// C/D: lane l holds col=l&15, rows (l>>4)*8 + r
__device__ __forceinline__ void cpos(int& mhi, int& col) {
  int l = threadIdx.x & 31; col = l & 15; mhi = (l >> 4) * 8;
}

// One 16x16 output tile. X [16 x 32*ksteps] (stride ldx halves),
// W swizzled: blocks [ntile][ktile][512].
__device__ __forceinline__ v8f gemm_tile(const h16* X, int ldx,
                                         const h16* W, int ksteps, int ntile) {
  const h16* wp = W + (size_t)ntile * ksteps * 512;
  v8f c = {};
  for (int k = 0; k < ksteps; ++k) {
    v16h a = frag_a(X + k * 32, ldx);
    v16h b = frag_bs(wp + (size_t)k * 512);
    c = __builtin_amdgcn_wmma_f32_16x16x32_f16(false, a, false, b,
                                               (short)0, c, false, false);
  }
  return c;
}

// ---------------------------------------------------------------------------
// Problem constants
// ---------------------------------------------------------------------------
#define BB    1024
#define NC    35
#define NA    2
#define NT    (BB*NC*NA)      // 71680 trees
#define NCARD (BB*NC)         // 35840 cards

// ---------------------------------------------------------------------------
// Weight repack: f32 [K x N] -> swizzled, zero-padded f16 fragment blocks
// dst layout: [nt][kt][col 0..15][slot 16h+s: kin = 8h+s (s<8), 8+8h+s (s>=8)]  (512 halves per block)
// ---------------------------------------------------------------------------
__global__ void pack_w(const float* src, h16* dst, int K, int N, int Kp, int Np) {
  int idx = blockIdx.x * blockDim.x + threadIdx.x;
  if (idx >= Kp * Np) return;
  int kin = ((idx & 8) << 1) | (((idx >> 4) & 1) << 3) | (idx & 7);
  int col = (idx >> 5) & 15;
  int blk = idx >> 9;
  int kts = Kp >> 5;
  int kt = blk % kts, nt = blk / kts;
  int k = kt * 32 + kin, n = nt * 16 + col;
  float v = (k < K && n < N) ? src[(size_t)k * N + n] : 0.0f;
  dst[idx] = (h16)v;
}

// ---------------------------------------------------------------------------
// Fused ability-tree encoder. 16 trees / block, 6 waves (N=96 -> 6 tiles).
// ---------------------------------------------------------------------------
__global__ __launch_bounds__(192)
void tree_kernel(const int* trig, const int* act, const int* targ, const int* opid,
                 const float* Etrig, const float* Eeff, const float* Etarg,
                 const float* Eop,
                 const h16* Wn1, const float* bn1,
                 const h16* Wn2, const float* bn2,
                 const h16* Wlf, const float* blf,
                 h16* root_out)
{
  __shared__ __attribute__((aligned(32))) h16   prim[7][16][96];
  __shared__ __attribute__((aligned(32))) h16   xa[16][192];
  __shared__ __attribute__((aligned(32))) h16   hbuf[16][96];
  __shared__ float acc[2][16][96];
  __shared__ float o1sum[16][96];
  __shared__ int   ops[7][16];
  __shared__ __attribute__((aligned(256))) unsigned char tdm_scratch[2048];

  int tid = threadIdx.x, wave = tid >> 5;
  int t0 = blockIdx.x * 16;

#if __has_builtin(__builtin_amdgcn_tensor_load_to_lds)
  // Tensor Data Mover: stream first 2KB of the swizzled node1 weights into an
  // LDS scratch tile (D# per ISA 8.3/8.4: count=1, type=2, 2D tile 512x1 dwords)
  // while the gather code below runs. Tracked by TENSORcnt.
  if (tid == 0) {
    unsigned lds_base = (unsigned)(size_t)(void*)&tdm_scratch[0];
    unsigned long long ga = (unsigned long long)(size_t)Wn1;
    u32x4 g0;
    g0[0] = 1u;                                              // count=1 (user D#)
    g0[1] = lds_base;                                        // lds_addr
    g0[2] = (unsigned)ga;                                    // global_addr[31:0]
    g0[3] = (unsigned)((ga >> 32) & 0x1FFFFFFu) | (2u << 30);// addr[56:32]|type=2
    unsigned td0 = 512u;                                     // dwords
    i32x8 g1;
    g1[0] = (int)(2u << 16);                  // data_size=4B
    g1[1] = (int)((td0 & 0xFFFFu) << 16);     // tensor_dim0[15:0]
    g1[2] = (int)((td0 >> 16) | (1u << 16));  // tensor_dim0[31:16] | tensor_dim1=1
    g1[3] = (int)(td0 << 16);                 // tile_dim0 = 512
    g1[4] = 1;                                // tile_dim1 = 1
    g1[5] = (int)td0;                         // tensor_dim0_stride lo
    g1[6] = 0; g1[7] = 0;
    i32x4 g2 = {0, 0, 0, 0};
    i32x4 g3 = {0, 0, 0, 0};
#if __clang_major__ >= 23
    i32x8 g4 = {0, 0, 0, 0, 0, 0, 0, 0};
    __builtin_amdgcn_tensor_load_to_lds(g0, g1, g2, g3, g4, 0);
#else
    __builtin_amdgcn_tensor_load_to_lds(g0, g1, g2, g3, 0);
#endif
  }
#endif

  // warm L2 / WGP$ for the weight blocks (global_prefetch_b8)
  __builtin_prefetch(Wn1 + (size_t)tid * 96, 0, 1);
  __builtin_prefetch(Wn2 + (size_t)tid * 48, 0, 1);
  __builtin_prefetch(Wlf + (size_t)tid * 48, 0, 1);

  for (int i = tid; i < 7 * 16; i += 192) {
    int j = i / 16, m = i % 16;
    ops[j][m] = opid[(size_t)(t0 + m) * 7 + j];
  }
  for (int i = tid; i < 7 * 16 * 96; i += 192) {
    int col = i % 96; int m = (i / 96) % 16; int j = i / (96 * 16);
    size_t idx = (size_t)(t0 + m) * 7 + j;
    float v = Etrig[trig[idx] * 96 + col] + Eeff[act[idx] * 96 + col]
            + Etarg[targ[idx] * 96 + col];
    prim[j][m][col] = (h16)v;
  }
  for (int i = tid; i < 2 * 16 * 96; i += 192) ((float*)acc)[i] = 0.f;
  for (int i = tid; i < 16 * 96; i += 192)     ((float*)o1sum)[i] = 0.f;
#if __has_builtin(__builtin_amdgcn_tensor_load_to_lds)
  __builtin_amdgcn_s_wait_tensorcnt(0);   // TDM scratch tile landed
#endif
  __syncthreads();

  int mhi, col; cpos(mhi, col);
  int nt = wave;                    // one N-tile of 16 per wave

  // ---- leaves (nodes 3..6): where(op==0, relu(leaf(p)), node2(relu(node1)))
  for (int j = 3; j < 7; ++j) {
    for (int i = tid; i < 16 * 192; i += 192) {
      int m = i / 192, c = i % 192;
      xa[m][c] = (c < 96) ? (h16)Eop[ops[j][m] * 96 + c] : prim[j][m][c - 96];
    }
    __syncthreads();
    v8f c1 = gemm_tile(&xa[0][0], 192, Wn1, 6, nt);
    float b1v = bn1[nt * 16 + col];
#pragma unroll
    for (int r = 0; r < 8; ++r) {
      float v = c1[r] + b1v; hbuf[mhi + r][nt * 16 + col] = (h16)(v > 0.f ? v : 0.f);
    }
    __syncthreads();
    v8f c2 = gemm_tile(&hbuf[0][0], 96, Wn2, 3, nt);
    v8f cl = gemm_tile(&prim[j][0][0], 96, Wlf, 3, nt);
    float b2v = bn2[nt * 16 + col], blv = blf[nt * 16 + col];
    int pair = (j - 3) >> 1;
#pragma unroll
    for (int r = 0; r < 8; ++r) {
      int m = mhi + r;
      float nbv = c2[r] + b2v;
      float lfv = cl[r] + blv; if (lfv < 0.f) lfv = 0.f;
      acc[pair][m][nt * 16 + col] += (ops[j][m] == 0) ? lfv : nbv;
    }
    __syncthreads();
  }

  // ---- inner nodes 1,2: node_mlp(op, prim + pairsum)
  for (int j = 1; j < 3; ++j) {
    for (int i = tid; i < 16 * 192; i += 192) {
      int m = i / 192, c = i % 192;
      xa[m][c] = (c < 96) ? (h16)Eop[ops[j][m] * 96 + c]
                          : (h16)((float)prim[j][m][c - 96] + acc[j - 1][m][c - 96]);
    }
    __syncthreads();
    v8f c1 = gemm_tile(&xa[0][0], 192, Wn1, 6, nt);
    float b1v = bn1[nt * 16 + col];
#pragma unroll
    for (int r = 0; r < 8; ++r) {
      float v = c1[r] + b1v; hbuf[mhi + r][nt * 16 + col] = (h16)(v > 0.f ? v : 0.f);
    }
    __syncthreads();
    v8f c2 = gemm_tile(&hbuf[0][0], 96, Wn2, 3, nt);
    float b2v = bn2[nt * 16 + col];
#pragma unroll
    for (int r = 0; r < 8; ++r) o1sum[mhi + r][nt * 16 + col] += c2[r] + b2v;
    __syncthreads();
  }

  // ---- root: node_mlp(op0, prim0 + out1.sum)
  for (int i = tid; i < 16 * 192; i += 192) {
    int m = i / 192, c = i % 192;
    xa[m][c] = (c < 96) ? (h16)Eop[ops[0][m] * 96 + c]
                        : (h16)((float)prim[0][m][c - 96] + o1sum[m][c - 96]);
  }
  __syncthreads();
  v8f c1 = gemm_tile(&xa[0][0], 192, Wn1, 6, nt);
  float b1v = bn1[nt * 16 + col];
#pragma unroll
  for (int r = 0; r < 8; ++r) {
    float v = c1[r] + b1v; hbuf[mhi + r][nt * 16 + col] = (h16)(v > 0.f ? v : 0.f);
  }
  __syncthreads();
  v8f c2 = gemm_tile(&hbuf[0][0], 96, Wn2, 3, nt);
  float b2v = bn2[nt * 16 + col];
#pragma unroll
  for (int r = 0; r < 8; ++r)
    root_out[(size_t)(t0 + mhi + r) * 96 + nt * 16 + col] = (h16)(c2[r] + b2v);
}

// ---------------------------------------------------------------------------
// ability(apool) + type/stats + LayerNorm + proj -> cards (f16, stride 96)
// ---------------------------------------------------------------------------
__global__ __launch_bounds__(192)
void card_kernel(const h16* root, const float* atk, const float* df,
                 const int* type_ids, const float* typeW, const float* typeB,
                 const h16* Wap, const float* bap,
                 const float* lng, const float* lnb,
                 const h16* Wpj, const float* bpj,
                 h16* cards)
{
  __shared__ __attribute__((aligned(32))) h16   rsum[16][96];
  __shared__ float raw[16][96];
  __shared__ __attribute__((aligned(32))) h16   nrm[16][96];
  int tid = threadIdx.x, wave = tid >> 5;
  int c0 = blockIdx.x * 16;

  __builtin_prefetch(Wap + (size_t)tid * 32, 0, 1);
  __builtin_prefetch(Wpj + (size_t)tid * 48, 0, 1);

  for (int i = tid; i < 16 * 96; i += 192) {
    int m = i / 96, c = i % 96;
    size_t t = (size_t)(c0 + m) * 2;
    rsum[m][c] = (h16)((float)root[t * 96 + c] + (float)root[(t + 1) * 96 + c]);
  }
  __syncthreads();
  int mhi, col; cpos(mhi, col);
  if (wave < 4) {   // apool 96 -> 64, relu
    v8f ca = gemm_tile(&rsum[0][0], 96, Wap, 3, wave);
    float bb = bap[wave * 16 + col];
#pragma unroll
    for (int r = 0; r < 8; ++r) {
      float v = ca[r] + bb; raw[mhi + r][18 + wave * 16 + col] = v > 0.f ? v : 0.f;
    }
  }
  for (int i = tid; i < 16 * 18; i += 192) {   // stats + type_vec
    int m = i / 18, c = i % 18, ci = c0 + m;
    float v;
    if (c == 0)      v = atk[ci] * 0.1f;
    else if (c == 1) v = df[ci] * 0.1f;
    else             v = typeW[type_ids[ci] * 16 + (c - 2)] + typeB[c - 2];
    raw[m][c] = v;
  }
  __syncthreads();
  if (tid < 16) {   // LayerNorm over 82 feats
    int m = tid;
    float mu = 0.f; for (int c = 0; c < 82; ++c) mu += raw[m][c];
    mu /= 82.f;
    float var = 0.f;
    for (int c = 0; c < 82; ++c) { float d = raw[m][c] - mu; var += d * d; }
    var /= 82.f;
    float inv = rsqrtf(var + 1e-5f);
    for (int c = 0; c < 96; ++c)
      nrm[m][c] = (h16)((c < 82) ? ((raw[m][c] - mu) * inv * lng[c] + lnb[c]) : 0.f);
  }
  __syncthreads();
  v8f cp = gemm_tile(&nrm[0][0], 96, Wpj, 3, wave);   // proj 82->82 (padded)
  int n = wave * 16 + col;
  float bb = (n < 82) ? bpj[n] : 0.f;
#pragma unroll
  for (int r = 0; r < 8; ++r) {
    float v = cp[r] + bb;
    cards[(size_t)(c0 + mhi + r) * 96 + n] = (h16)(n < 82 ? v : 0.f);
  }
}

// ---------------------------------------------------------------------------
// Generic fused 2-layer MLP: Y = act2( relu(X@W1+b1) @ W2 + b2 )
// ---------------------------------------------------------------------------
__global__ __launch_bounds__(128)
void mlp2_kernel(const h16* X, int ldx, int k1steps,
                 const h16* W1, const float* b1, int N1,
                 const h16* W2, const float* b2, int N2, int n2valid, int relu2,
                 h16* Y, int ldy)
{
  __shared__ __attribute__((aligned(32))) h16 hbuf[16][128];
  int tid = threadIdx.x, wave = tid >> 5;
  int m0 = blockIdx.x * 16;
  int mhi, col; cpos(mhi, col);
  __builtin_prefetch(W1 + (size_t)tid * 64, 0, 1);
  __builtin_prefetch(W2 + (size_t)tid * 64, 0, 1);
  const h16* Xt = X + (size_t)m0 * ldx;
  for (int nt = wave; nt < N1 / 16; nt += 4) {
    v8f c = gemm_tile(Xt, ldx, W1, k1steps, nt);
    float bb = b1[nt * 16 + col];
#pragma unroll
    for (int r = 0; r < 8; ++r) {
      float v = c[r] + bb; hbuf[mhi + r][nt * 16 + col] = (h16)(v > 0.f ? v : 0.f);
    }
  }
  __syncthreads();
  for (int nt = wave; nt < N2 / 16; nt += 4) {
    v8f c = gemm_tile(&hbuf[0][0], 128, W2, N1 / 32, nt);
    int n = nt * 16 + col;
    float bb = (n < n2valid) ? b2[n] : 0.f;
#pragma unroll
    for (int r = 0; r < 8; ++r) {
      float v = c[r] + bb;
      if (relu2 && v < 0.f) v = 0.f;
      Y[(size_t)(m0 + mhi + r) * ldy + n] = (h16)v;
    }
  }
}

// ---------------------------------------------------------------------------
// zone sums over card slices
// ---------------------------------------------------------------------------
__global__ void zsum_kernel(const h16* percard, h16* zsum) {
  int idx = blockIdx.x * blockDim.x + threadIdx.x;
  if (idx >= BB * 5 * 64) return;
  int col = idx % 64, z = (idx / 64) % 5, b = idx / (64 * 5);
  const int off[5] = {0, 7, 15, 22, 30};
  const int len[5] = {7, 8, 7, 8, 5};
  float s = 0.f;
  for (int i = 0; i < len[z]; ++i)
    s += (float)percard[((size_t)b * NC + off[z] + i) * 64 + col];
  zsum[idx] = (h16)s;
}

// ---------------------------------------------------------------------------
// Relation net: block = (b, q); M-tile = 14 k-partners (padded to 16).
// relsum[b*14+q][:] = sum_k rel2(relu(rel1([q_card, k_card])))
// ---------------------------------------------------------------------------
__global__ __launch_bounds__(128)
void rel_kernel(const h16* cards, const h16* Wr1, const float* br1,
                const h16* Wr2, const float* br2, h16* relsum)
{
  __shared__ __attribute__((aligned(32))) h16   xa[16][192];
  __shared__ __attribute__((aligned(32))) h16   hbuf[16][64];
  __shared__ float ct[16][64];
  int tid = threadIdx.x, wave = tid >> 5;
  int b = blockIdx.x / 14, q = blockIdx.x % 14;
  int qc = b * NC + (q < 7 ? q : 8 + q);     // BF_SELF or BF_OPP card
  __builtin_prefetch(Wr1 + (size_t)tid * 96, 0, 1);
  __builtin_prefetch(Wr2 + (size_t)tid * 64, 0, 1);
  for (int i = tid; i < 16 * 192; i += 128) {
    int m = i / 192, c = i % 192;
    h16 v = (h16)0.f;
    if (m < 14) {
      int kc = b * NC + (m < 7 ? m : 8 + m);
      if (c < 82)       v = cards[(size_t)qc * 96 + c];
      else if (c < 164) v = cards[(size_t)kc * 96 + (c - 82)];
    }
    xa[m][c] = v;
  }
  __syncthreads();
  int mhi, col; cpos(mhi, col);
  v8f c1 = gemm_tile(&xa[0][0], 192, Wr1, 6, wave);
  float bb = br1[wave * 16 + col];
#pragma unroll
  for (int r = 0; r < 8; ++r) {
    float v = c1[r] + bb; hbuf[mhi + r][wave * 16 + col] = (h16)(v > 0.f ? v : 0.f);
  }
  __syncthreads();
  v8f c2 = gemm_tile(&hbuf[0][0], 64, Wr2, 2, wave);
#pragma unroll
  for (int r = 0; r < 8; ++r) ct[mhi + r][wave * 16 + col] = c2[r];
  __syncthreads();
  if (tid < 64) {            // sum the 14 valid rows (+14*bias)
    float s = 14.f * br2[tid];
    for (int m = 0; m < 14; ++m) s += ct[m][tid];
    relsum[(size_t)blockIdx.x * 64 + tid] = (h16)s;
  }
}

// ---------------------------------------------------------------------------
// ctx: enr = relu(ctx([card, rel])).sum over 7 battlefield cards, per side
// ---------------------------------------------------------------------------
__global__ __launch_bounds__(128)
void ctx_kernel(const h16* cards, const h16* relsum,
                const h16* Wc, const float* bc, h16* enr)
{
  __shared__ __attribute__((aligned(32))) h16   xa[16][160];
  __shared__ float ct[16][64];
  int tid = threadIdx.x, wave = tid >> 5;
  int b = blockIdx.x >> 1, s = blockIdx.x & 1;
  __builtin_prefetch(Wc + (size_t)tid * 80, 0, 1);
  for (int i = tid; i < 16 * 160; i += 128) {
    int m = i / 160, c = i % 160;
    h16 v = (h16)0.f;
    if (m < 7) {
      int card = b * NC + (s == 0 ? m : 15 + m);
      int rrow = b * 14 + s * 7 + m;
      if (c < 82)       v = cards[(size_t)card * 96 + c];
      else if (c < 146) v = relsum[(size_t)rrow * 64 + (c - 82)];
    }
    xa[m][c] = v;
  }
  __syncthreads();
  int mhi, col; cpos(mhi, col);
  v8f c1 = gemm_tile(&xa[0][0], 160, Wc, 5, wave);
  float bb = bc[wave * 16 + col];
#pragma unroll
  for (int r = 0; r < 8; ++r) {
    float v = c1[r] + bb; ct[mhi + r][wave * 16 + col] = v > 0.f ? v : 0.f;
  }
  __syncthreads();
  if (tid < 64) {
    float s2 = 0.f;
    for (int m = 0; m < 7; ++m) s2 += ct[m][tid];
    enr[(size_t)blockIdx.x * 64 + tid] = (h16)s2;
  }
}

// ---------------------------------------------------------------------------
// small gathers / finalize
// ---------------------------------------------------------------------------
__global__ void state_kernel(const float* scalars, const h16* zones, h16* state) {
  int idx = blockIdx.x * blockDim.x + threadIdx.x;
  if (idx >= BB * 352) return;
  int c = idx % 352, b = idx / 352;
  h16 v = (h16)0.f;
  if (c < 7)        v = (h16)scalars[b * 7 + c];
  else if (c < 327) v = zones[(size_t)b * 320 + (c - 7)];
  state[idx] = v;
}

__global__ void comb_kernel(const h16* enr, const h16* prep,
                            const float* scalars, h16* comb) {
  int idx = blockIdx.x * blockDim.x + threadIdx.x;
  if (idx >= BB * 224) return;
  int c = idx % 224, b = idx / 224;
  h16 v = (h16)0.f;
  if (c < 64)       v = enr[(size_t)(b * 2) * 64 + c];
  else if (c < 128) v = enr[(size_t)(b * 2 + 1) * 64 + (c - 64)];
  else if (c < 192) v = prep[(size_t)b * 64 + (c - 128)];
  else if (c < 199) v = (h16)scalars[b * 7 + (c - 192)];
  comb[idx] = v;
}

__global__ void fin_kernel(const h16* vhout, float* out) {
  int i = blockIdx.x * blockDim.x + threadIdx.x;
  if (i < BB) out[i] = tanhf((float)vhout[(size_t)i * 16]);
}

// ---------------------------------------------------------------------------
// Host orchestration
// ---------------------------------------------------------------------------
extern "C" void kernel_launch(void* const* d_in, const int* in_sizes, int n_in,
                              void* d_out, int out_size, void* d_ws, size_t ws_size,
                              hipStream_t stream) {
  (void)in_sizes; (void)n_in; (void)out_size; (void)ws_size;
  // d_in layout: inputs in dict order, then params leaves in sorted-key pytree
  // order (tuples expanded as (W, b)).
  const float* atk      = (const float*)d_in[0];
  const float* df       = (const float*)d_in[1];
  const float* scalars  = (const float*)d_in[2];
  const int*   type_ids = (const int*)d_in[3];
  const int*   trig_ids = (const int*)d_in[4];
  const int*   act_ids  = (const int*)d_in[5];
  const int*   targ_ids = (const int*)d_in[6];
  const int*   op_ids   = (const int*)d_in[7];
  const float* E_eff  = (const float*)d_in[8];
  const float* E_op   = (const float*)d_in[9];
  const float* E_targ = (const float*)d_in[10];
  const float* E_trig = (const float*)d_in[11];
  const float* apool_w = (const float*)d_in[12]; const float* apool_b = (const float*)d_in[13];
  const float* ctx_w   = (const float*)d_in[14]; const float* ctx_b   = (const float*)d_in[15];
  const float* leaf_w  = (const float*)d_in[16]; const float* leaf_b  = (const float*)d_in[17];
  const float* ln_g    = (const float*)d_in[18]; const float* ln_b    = (const float*)d_in[19];
  const float* node1_w = (const float*)d_in[20]; const float* node1_b = (const float*)d_in[21];
  const float* node2_w = (const float*)d_in[22]; const float* node2_b = (const float*)d_in[23];
  const float* phi1_w  = (const float*)d_in[24]; const float* phi1_b  = (const float*)d_in[25];
  const float* phi2_w  = (const float*)d_in[26]; const float* phi2_b  = (const float*)d_in[27];
  const float* proj_w  = (const float*)d_in[28]; const float* proj_b  = (const float*)d_in[29];
  const float* rel1_w  = (const float*)d_in[30]; const float* rel1_b  = (const float*)d_in[31];
  const float* rel2_w  = (const float*)d_in[32]; const float* rel2_b  = (const float*)d_in[33];
  const float* rho1_w  = (const float*)d_in[34]; const float* rho1_b  = (const float*)d_in[35];
  const float* rho2_w  = (const float*)d_in[36]; const float* rho2_b  = (const float*)d_in[37];
  const float* tw1_w   = (const float*)d_in[38]; const float* tw1_b   = (const float*)d_in[39];
  const float* tw2_w   = (const float*)d_in[40]; const float* tw2_b   = (const float*)d_in[41];
  const float* type_w  = (const float*)d_in[42]; const float* type_b  = (const float*)d_in[43];
  const float* vh1_w   = (const float*)d_in[44]; const float* vh1_b   = (const float*)d_in[45];
  const float* vh2_w   = (const float*)d_in[46]; const float* vh2_b   = (const float*)d_in[47];

  // Workspace bump allocator (f16 elements, 256B aligned)
  size_t off = 0;
  auto alloc_h = [&](size_t n) -> h16* {
    h16* p = (h16*)((char*)d_ws + off);
    off += ((n * sizeof(h16) + 255) / 256) * 256;
    return p;
  };
  // packed (swizzled) f16 weights
  h16* Pn1 = alloc_h(192 * 96);  h16* Pn2 = alloc_h(96 * 96);  h16* Plf = alloc_h(96 * 96);
  h16* Pap = alloc_h(96 * 64);   h16* Ppj = alloc_h(96 * 96);
  h16* Pp1 = alloc_h(96 * 64);   h16* Pp2 = alloc_h(64 * 64);
  h16* Pr1 = alloc_h(64 * 64);   h16* Pr2 = alloc_h(64 * 64);
  h16* Prl1 = alloc_h(192 * 64); h16* Prl2 = alloc_h(64 * 64);
  h16* Pcx = alloc_h(160 * 64);
  h16* Pt1 = alloc_h(352 * 128); h16* Pt2 = alloc_h(128 * 64);
  h16* Pv1 = alloc_h(224 * 128); h16* Pv2 = alloc_h(128 * 16);
  // activations
  h16* root    = alloc_h((size_t)NT * 96);
  h16* cards   = alloc_h((size_t)NCARD * 96);
  h16* percard = alloc_h((size_t)NCARD * 64);
  h16* zsum    = alloc_h((size_t)BB * 5 * 64);
  h16* zones   = alloc_h((size_t)BB * 5 * 64);
  h16* relsum  = alloc_h((size_t)BB * 14 * 64);
  h16* enr     = alloc_h((size_t)BB * 2 * 64);
  h16* state   = alloc_h((size_t)BB * 352);
  h16* prep    = alloc_h((size_t)BB * 64);
  h16* comb    = alloc_h((size_t)BB * 224);
  h16* vhout   = alloc_h((size_t)BB * 16);

  auto pk = [&](const float* src, h16* dst, int K, int N, int Kp, int Np) {
    int tot = Kp * Np;
    pack_w<<<(tot + 255) / 256, 256, 0, stream>>>(src, dst, K, N, Kp, Np);
  };
  pk(node1_w, Pn1, 192, 96, 192, 96);
  pk(node2_w, Pn2, 96, 96, 96, 96);
  pk(leaf_w,  Plf, 96, 96, 96, 96);
  pk(apool_w, Pap, 96, 64, 96, 64);
  pk(proj_w,  Ppj, 82, 82, 96, 96);
  pk(phi1_w,  Pp1, 82, 64, 96, 64);
  pk(phi2_w,  Pp2, 64, 64, 64, 64);
  pk(rho1_w,  Pr1, 64, 64, 64, 64);
  pk(rho2_w,  Pr2, 64, 64, 64, 64);
  pk(rel1_w,  Prl1, 164, 64, 192, 64);
  pk(rel2_w,  Prl2, 64, 64, 64, 64);
  pk(ctx_w,   Pcx, 146, 64, 160, 64);
  pk(tw1_w,   Pt1, 327, 128, 352, 128);
  pk(tw2_w,   Pt2, 128, 64, 128, 64);
  pk(vh1_w,   Pv1, 199, 128, 224, 128);
  pk(vh2_w,   Pv2, 128, 1, 128, 16);

  // 1. ability trees  (dominant compute: WMMA)
  tree_kernel<<<NT / 16, 192, 0, stream>>>(trig_ids, act_ids, targ_ids, op_ids,
      E_trig, E_eff, E_targ, E_op, Pn1, node1_b, Pn2, node2_b, Plf, leaf_b, root);
  // 2. per-card features: apool + type + LN + proj
  card_kernel<<<NCARD / 16, 192, 0, stream>>>(root, atk, df, type_ids,
      type_w, type_b, Pap, apool_b, ln_g, ln_b, Ppj, proj_b, cards);
  // 3. zone deep-set phi
  mlp2_kernel<<<NCARD / 16, 128, 0, stream>>>(cards, 96, 3, Pp1, phi1_b, 64,
      Pp2, phi2_b, 64, 64, 0, percard, 64);
  zsum_kernel<<<(BB * 5 * 64 + 255) / 256, 256, 0, stream>>>(percard, zsum);
  mlp2_kernel<<<BB * 5 / 16, 128, 0, stream>>>(zsum, 64, 2, Pr1, rho1_b, 64,
      Pr2, rho2_b, 64, 64, 0, zones, 64);
  // 4. relation net over battlefield pairs
  rel_kernel<<<BB * 14, 128, 0, stream>>>(cards, Prl1, rel1_b, Prl2, rel2_b, relsum);
  ctx_kernel<<<BB * 2, 128, 0, stream>>>(cards, relsum, Pcx, ctx_b, enr);
  // 5. tower
  state_kernel<<<(BB * 352 + 255) / 256, 256, 0, stream>>>(scalars, zones, state);
  mlp2_kernel<<<BB / 16, 128, 0, stream>>>(state, 352, 11, Pt1, tw1_b, 128,
      Pt2, tw2_b, 64, 64, 1, prep, 64);
  // 6. value head
  comb_kernel<<<(BB * 224 + 255) / 256, 256, 0, stream>>>(enr, prep, scalars, comb);
  mlp2_kernel<<<BB / 16, 128, 0, stream>>>(comb, 224, 7, Pv1, vh1_b, 128,
      Pv2, vh2_b, 16, 1, 0, vhout, 16);
  fin_kernel<<<(BB + 255) / 256, 256, 0, stream>>>(vhout, (float*)d_out);
}
